// MAGNET_19353122636197
// MI455X (gfx1250) — matrix operand order repaired and checked by reading, not compiled, not run on hardware
//
#include <hip/hip_runtime.h>
#include <hip/hip_bf16.h>
#include <math.h>

// ---------------------------------------------------------------------------
// Problem constants (match reference)
// ---------------------------------------------------------------------------
#define Bn   2
#define Vn   4
#define Cn   64
#define FDn  256
#define Hn   120
#define Wn   160
#define Kn   5
#define UPn  4
#define HWn  (Hn * Wn)        // 19200
#define BHWn (Bn * HWn)       // 38400
#define HOUT (Hn * UPn)       // 480
#define WOUT (Wn * UPn)       // 640
#define GINC (FDn + Kn)       // 261 channels of gnet input

typedef __attribute__((ext_vector_type(16))) _Float16 v16h;
typedef __attribute__((ext_vector_type(8)))  _Float16 v8h;
typedef __attribute__((ext_vector_type(8)))  float    v8f;

// ---------------------------------------------------------------------------
// Weight packing: w[Cout][Cin*ks*ks] (f32) -> fragment-ordered f16:
//   Bpack[(k/32) * N_alloc + n][k%32]
// so a wave32 B-fragment (16 consecutive K-halves for one column n) is a
// contiguous 32-byte run -> two global_load_b128 per lane, no LDS staging.
// Rows k >= K_real and cols n >= Cout are zero.
// ---------------------------------------------------------------------------
__global__ __launch_bounds__(256) void pack_weights_k(
    const float* __restrict__ w, _Float16* __restrict__ Bpack,
    int Cout, int K_real, int K_pad, int N_alloc)
{
  int i = blockIdx.x * blockDim.x + threadIdx.x;
  int total = K_pad * N_alloc;
  if (i >= total) return;
  int k = i / N_alloc;
  int n = i - k * N_alloc;
  _Float16 v = (_Float16)0.f;
  if (k < K_real && n < Cout) v = (_Float16)w[(size_t)n * K_real + k];
  Bpack[((size_t)(k >> 5) * N_alloc + n) * 32 + (k & 31)] = v;
}

// x_d3 (f32, planar B x FD x H x W) -> gin channels [5..260] (f16)
__global__ __launch_bounds__(256) void pack_xd3_k(
    const float* __restrict__ x, _Float16* __restrict__ gin)
{
  int i = blockIdx.x * blockDim.x + threadIdx.x;
  int total = Bn * FDn * HWn;
  if (i >= total) return;
  int b  = i / (FDn * HWn);
  int r  = i - b * (FDn * HWn);
  int c  = r / HWn;
  int hw = r - c * HWn;
  gin[(size_t)b * GINC * HWn + (size_t)(Kn + c) * HWn + hw] = (_Float16)x[i];
}

// ---------------------------------------------------------------------------
// Cost volume: one thread per (b, k, h, w).  Projects the depth hypothesis
// into each neighbor view, bilinear-samples 64 feature + 2 GMM channels
// (L2-resident gathers: nghbr_feat is ~39 MB << 192 MB L2), accumulates the
// weighted similarity, writes the result as f16 directly into gin ch [0..4].
// ---------------------------------------------------------------------------
__global__ __launch_bounds__(256) void cost_volume_k(
    const float* __restrict__ ref_feat, const float* __restrict__ nf,
    const float* __restrict__ ngmm, const float* __restrict__ prev,
    const float* __restrict__ Rs, const float* __restrict__ ts,
    const float* __restrict__ is_valid, const float* __restrict__ intrins,
    _Float16* __restrict__ gin)
{
  int i = blockIdx.x * blockDim.x + threadIdx.x;
  if (i >= Bn * Kn * HWn) return;
  int b  = i / (Kn * HWn);
  int r  = i - b * (Kn * HWn);
  int k  = r / HWn;
  int hw = r - k * HWn;
  int h  = hw / Wn;
  int w  = hw - h * Wn;

  float mu = prev[b * 2 * HWn + hw];
  float sg = prev[b * 2 * HWn + HWn + hw];
  float kl = (k == 0) ? -1.9192f : (k == 1) ? -0.5456f : (k == 2) ? 0.f
           : (k == 3) ? 0.5456f : 1.9192f;
  float d = mu + sg * kl;

  float fx = intrins[b * 4 + 0], fy = intrins[b * 4 + 1];
  float cx = intrins[b * 4 + 2], cy = intrins[b * 4 + 3];
  float X0 = d * ((float)w - cx) / fx;
  float X1 = d * ((float)h - cy) / fy;
  float X2 = d;

  const float* ref = ref_feat + (size_t)b * Cn * HWn + hw;

  float num = 0.f, den = 0.f;
  for (int v = 0; v < Vn; ++v) {
    const float* R = Rs + (size_t)(b * Vn + v) * 9;
    const float* t = ts + (size_t)(b * Vn + v) * 3;
    float xp = R[0] * X0 + R[1] * X1 + R[2] * X2 + t[0];
    float yp = R[3] * X0 + R[4] * X1 + R[5] * X2 + t[1];
    float zp = R[6] * X0 + R[7] * X1 + R[8] * X2 + t[2];
    if (zp <= 0.001f) continue;                       // weight == 0 anyway
    float up = fx * xp / zp + cx;
    float vp = fy * yp / zp + cy;
    if (!(up >= 0.f && up <= (float)(Wn - 1) &&
          vp >= 0.f && vp <= (float)(Hn - 1))) continue;

    float u0f = floorf(up), v0f = floorf(vp);
    float du = up - u0f, dv = vp - v0f;
    int u0 = (int)u0f, v0 = (int)v0f;
    int u1 = u0 + 1,   v1 = v0 + 1;
    float w00 = (1.f - du) * (1.f - dv);
    float w10 = du * (1.f - dv);
    float w01 = (1.f - du) * dv;
    float w11 = du * dv;
    float m10 = (u1 < Wn) ? 1.f : 0.f;   // per-tap validity (matches reference)
    float m01 = (v1 < Hn) ? 1.f : 0.f;
    w10 *= m10; w01 *= m01; w11 *= m10 * m01;
    int u1c = (u1 < Wn) ? u1 : (Wn - 1);
    int v1c = (v1 < Hn) ? v1 : (Hn - 1);
    int o00 = v0  * Wn + u0;
    int o10 = v0  * Wn + u1c;
    int o01 = v1c * Wn + u0;
    int o11 = v1c * Wn + u1c;

    const float* fbase = nf + (size_t)(b * Vn + v) * Cn * HWn;
    float sim = 0.f;
    #pragma unroll 4
    for (int c = 0; c < Cn; ++c) {
      const float* f = fbase + (size_t)c * HWn;
      float s = w00 * f[o00] + w10 * f[o10] + w01 * f[o01] + w11 * f[o11];
      sim += ref[(size_t)c * HWn] * s;
    }
    sim *= 0.125f;                                    // 1/sqrt(64)

    const float* g0 = ngmm + (size_t)(b * Vn + v) * 2 * HWn;
    const float* g1 = g0 + HWn;
    float mun = w00 * g0[o00] + w10 * g0[o10] + w01 * g0[o01] + w11 * g0[o11];
    float sgn = w00 * g1[o00] + w10 * g1[o10] + w01 * g1[o01] + w11 * g1[o11];
    if (fabsf(zp - mun) < 5.0f * fmaxf(sgn, 1e-6f)) {
      float wv = is_valid[b * Vn + v];
      num += wv * sim;
      den += wv;
    }
  }
  gin[(size_t)b * GINC * HWn + (size_t)k * HWn + hw] =
      (_Float16)(num / (den + 1e-6f));
}

// ---------------------------------------------------------------------------
// Implicit-GEMM conv (KSZ=1 or 3, pad KSZ/2) using wave32 WMMA.
//   Block = 256 threads = 8 waves, M-tile = 64 pixels, N-tile = 128.
//   Per 32-wide K-step:
//     * A tile (64x32 f16) staged in LDS: each thread gathers 8 im2col halves
//       into a v8h and stores with one ds_store_b128.
//     * B fragment loaded straight from fragment-ordered global weights
//       (L2-resident): two global_load_b128 per lane, no LDS, + prefetch.
//     * Each wave reuses its B fragment across 4 M sub-tiles ->
//       4x v_wmma_f32_16x16x32_f16 per K-step; A fragments are two
//       contiguous ds_load_b128 per sub-tile.
// ---------------------------------------------------------------------------
template <int KSZ>
__global__ __launch_bounds__(256) void wmma_conv_gemm(
    const _Float16* __restrict__ in, int in_bstride, int in_ch_off,
    const _Float16* __restrict__ Bpack, const float* __restrict__ bias,
    _Float16* __restrict__ out_h, float* __restrict__ out_f,
    int K_real, int K_pad, int N_real, int N_alloc, int do_relu)
{
  __shared__ __align__(16) _Float16 As[64 * 32];   // 4 KB

  const int t    = threadIdx.x;
  const int lane = t & 31;
  const int wave = t >> 5;
  const int m_base  = blockIdx.x * 64;
  const int n_block = blockIdx.y * 128;
  constexpr int pad_ = KSZ >> 1;
  constexpr int kk2  = KSZ * KSZ;

  // ---- per-thread fixed im2col row for the staging phase (hoisted) ----
  const int sm  = t >> 2;          // 0..63 : row within A tile
  const int sch = (t & 3) << 3;    // 0,8,16,24 : k-chunk within 32
  const int spix = m_base + sm;
  const int sb   = spix / HWn;
  const int shw  = spix - sb * HWn;
  const int sh   = shw / Wn;
  const int sw   = shw - sh * Wn;
  const _Float16* sin = in + (size_t)sb * in_bstride + (size_t)in_ch_off * HWn;

  // ---- fragment geometry (wave32; k contiguous per lane half, same order for A and B) ----
  const int mr  = lane & 15;
  const int kbA = (lane < 16) ? 0 : 16;   // A: halves [kbA..+15] (contiguous)
  const int kbB = (lane < 16) ? 0 : 16;   // B: halves [kbB..+15] (contiguous)
  const int ng  = n_block + (wave << 4) + mr;
  const bool bvalid = (ng < N_alloc);

  v8f accs[4] = {};

  for (int k0 = 0; k0 < K_pad; k0 += 32) {
    // ---- stage A: gather 8 halves, one ds_store_b128 ----
    {
      v8h tmp;
      #pragma unroll
      for (int jj = 0; jj < 8; ++jj) {
        int k = k0 + sch + jj;
        _Float16 val = (_Float16)0.f;
        if (k < K_real) {
          if constexpr (KSZ == 1) {
            val = sin[(size_t)k * HWn + shw];
          } else {
            int cin = k / kk2;
            int rr  = k - cin * kk2;
            int ky  = rr / KSZ;
            int kx  = rr - ky * KSZ;
            int hy  = sh + ky - pad_;
            int wx  = sw + kx - pad_;
            if (hy >= 0 && hy < Hn && wx >= 0 && wx < Wn)
              val = sin[(size_t)cin * HWn + hy * Wn + wx];
          }
        }
        tmp[jj] = val;
      }
      *reinterpret_cast<v8h*>(&As[sm * 32 + sch]) = tmp;
    }

    // ---- B fragment: two global_load_b128 from packed weights ----
    v16h bf = {};
    if (bvalid) {
      const _Float16* bp =
          Bpack + ((size_t)(k0 >> 5) * N_alloc + ng) * 32 + kbB;
      v8h b0 = *reinterpret_cast<const v8h*>(bp);
      v8h b1 = *reinterpret_cast<const v8h*>(bp + 8);
      bf = __builtin_shufflevector(b0, b1, 0, 1, 2, 3, 4, 5, 6, 7,
                                           8, 9, 10, 11, 12, 13, 14, 15);
      if (k0 + 32 < K_pad)
        __builtin_prefetch(bp + (size_t)N_alloc * 32, 0, 1);
    }
    __syncthreads();

    // ---- 4 M sub-tiles: 2x ds_load_b128 + WMMA each ----
    #pragma unroll
    for (int ms = 0; ms < 4; ++ms) {
      const _Float16* ap = &As[(ms * 16 + mr) * 32 + kbA];
      v8h a0 = *reinterpret_cast<const v8h*>(ap);
      v8h a1 = *reinterpret_cast<const v8h*>(ap + 8);
      v16h af = __builtin_shufflevector(a0, a1, 0, 1, 2, 3, 4, 5, 6, 7,
                                                8, 9, 10, 11, 12, 13, 14, 15);
      accs[ms] = __builtin_amdgcn_wmma_f32_16x16x32_f16(
          false, af, false, bf, (short)0, accs[ms], false, false);
    }
    __syncthreads();
  }

  // ---- epilogue: bias (+ReLU), store f16 planar and/or f32 planar ----
  if (ng < N_real) {
    float bval = bias ? bias[ng] : 0.f;
    #pragma unroll
    for (int ms = 0; ms < 4; ++ms) {
      #pragma unroll
      for (int r2 = 0; r2 < 8; ++r2) {
        int m   = ms * 16 + r2 + ((lane < 16) ? 0 : 8);  // C/D VGPR r -> M=r(+8)
        int pix = m_base + m;
        float v = accs[ms][r2] + bval;
        if (do_relu) v = fmaxf(v, 0.f);
        int b  = pix / HWn;
        int hw = pix - b * HWn;
        if (out_h) out_h[((size_t)b * N_real + ng) * HWn + hw] = (_Float16)v;
        if (out_f) out_f[((size_t)b * N_real + ng) * HWn + hw] = v;
      }
    }
  }
}

// ---------------------------------------------------------------------------
// gnet output head: 1x1 conv 128->2 (scalar dot), GMM update with ELU.
// ---------------------------------------------------------------------------
__global__ __launch_bounds__(256) void gnet_head_k(
    const _Float16* __restrict__ h3, const float* __restrict__ gw4,
    const float* __restrict__ gb4, const float* __restrict__ prev,
    float* __restrict__ pred)
{
  int i = blockIdx.x * blockDim.x + threadIdx.x;
  if (i >= BHWn) return;
  int b  = i / HWn;
  int hw = i - b * HWn;
  const _Float16* hp = h3 + (size_t)b * 128 * HWn + hw;
  float mu1 = 0.f, sg1 = 0.f;
  #pragma unroll 4
  for (int c = 0; c < 128; ++c) {
    float hv = (float)hp[(size_t)c * HWn];
    mu1 += hv * gw4[c];
    sg1 += hv * gw4[128 + c];
  }
  mu1 += gb4[0];
  sg1 += gb4[1];
  float mu0 = prev[b * 2 * HWn + hw];
  float s0  = prev[b * 2 * HWn + HWn + hw];
  float elu = (sg1 > 0.f) ? sg1 : (expf(sg1) - 1.f);
  pred[b * 2 * HWn + hw]       = mu0 + mu1 * s0;
  pred[b * 2 * HWn + HWn + hw] = (elu + 1.f + 1e-10f) * s0;
}

// ---------------------------------------------------------------------------
// Convex upsample: softmax over 9 mask logits per (b,h,w,i,j), applied to the
// 3x3 neighborhood of each iteration's predicted depth/sigma.  Writes d_out
// directly in (iter, b, chan, 480, 640) order.
// ---------------------------------------------------------------------------
__global__ __launch_bounds__(256) void upsample_k(
    const float* __restrict__ mask, const float* __restrict__ preds,
    float* __restrict__ out)
{
  int i = blockIdx.x * blockDim.x + threadIdx.x;
  int total = Bn * HWn * 16;
  if (i >= total) return;
  int b  = i / (HWn * 16);
  int r  = i - b * (HWn * 16);
  int hw = r >> 4;
  int ij = r & 15;
  int h  = hw / Wn;
  int w  = hw - h * Wn;
  int ii = ij >> 2;
  int jj = ij & 3;

  float m[9];
  float mx = -1e30f;
  #pragma unroll
  for (int n = 0; n < 9; ++n) {
    m[n] = mask[((size_t)b * 144 + (n * 16 + ij)) * HWn + hw];
    mx = fmaxf(mx, m[n]);
  }
  float s = 0.f;
  #pragma unroll
  for (int n = 0; n < 9; ++n) { m[n] = expf(m[n] - mx); s += m[n]; }
  float inv = 1.f / s;
  #pragma unroll
  for (int n = 0; n < 9; ++n) m[n] *= inv;

  for (int it = 0; it < 3; ++it) {
    const float* dp = preds + ((size_t)it * Bn + b) * 2 * HWn;
    for (int o = 0; o < 2; ++o) {
      const float* dd = dp + (size_t)o * HWn;
      float acc = 0.f;
      #pragma unroll
      for (int n = 0; n < 9; ++n) {
        int hh = h + n / 3 - 1;
        int ww = w + n % 3 - 1;
        float dv = (hh >= 0 && hh < Hn && ww >= 0 && ww < Wn)
                     ? dd[hh * Wn + ww] : 0.f;
        acc += m[n] * dv;
      }
      out[(((size_t)it * Bn + b) * 2 + o) * ((size_t)HOUT * WOUT) +
          (size_t)(h * UPn + ii) * WOUT + (w * UPn + jj)] = acc;
    }
  }
}

// ---------------------------------------------------------------------------
// Host orchestration
// ---------------------------------------------------------------------------
extern "C" void kernel_launch(void* const* d_in, const int* in_sizes, int n_in,
                              void* d_out, int out_size, void* d_ws,
                              size_t ws_size, hipStream_t stream)
{
  (void)in_sizes; (void)n_in; (void)out_size; (void)ws_size;

  const float* ref_feat   = (const float*)d_in[0];
  const float* nghbr_feat = (const float*)d_in[1];
  const float* ref_gmms   = (const float*)d_in[2];
  const float* nghbr_gmms = (const float*)d_in[3];
  const float* x_d3       = (const float*)d_in[4];
  const float* Rs         = (const float*)d_in[5];
  const float* ts         = (const float*)d_in[6];
  const float* is_valid   = (const float*)d_in[7];
  const float* intrins    = (const float*)d_in[8];
  const float* gw1 = (const float*)d_in[9],  *gb1 = (const float*)d_in[10];
  const float* gw2 = (const float*)d_in[11], *gb2 = (const float*)d_in[12];
  const float* gw3 = (const float*)d_in[13], *gb3 = (const float*)d_in[14];
  const float* gw4 = (const float*)d_in[15], *gb4 = (const float*)d_in[16];
  const float* mw1 = (const float*)d_in[17], *mb1 = (const float*)d_in[18];
  const float* mw2 = (const float*)d_in[19], *mb2 = (const float*)d_in[20];
  const float* mw3 = (const float*)d_in[21], *mb3 = (const float*)d_in[22];
  const float* mw4 = (const float*)d_in[23], *mb4 = (const float*)d_in[24];
  float* out = (float*)d_out;

  // ---- workspace carve-up (~61 MiB total) ----
  char* ws = (char*)d_ws;
  size_t off = 0;
  auto carve = [&](size_t bytes) {
    size_t o = off;
    off += (bytes + 255) & ~(size_t)255;
    return o;
  };
  _Float16* gin   = (_Float16*)(ws + carve((size_t)Bn * GINC * HWn * 2));
  _Float16* hA    = (_Float16*)(ws + carve((size_t)Bn * 128 * HWn * 2));
  _Float16* hB    = (_Float16*)(ws + carve((size_t)Bn * 128 * HWn * 2));
  float*    maskb = (float*)   (ws + carve((size_t)Bn * 144 * HWn * 4));
  float*    preds = (float*)   (ws + carve((size_t)3 * Bn * 2 * HWn * 4));

  const int KG1_REAL = GINC * 9;                    // 2349
  const int KG1_PAD  = ((KG1_REAL + 31) / 32) * 32; // 2368
  const int KM1_REAL = FDn * 9;                     // 2304 (already /32)

  _Float16* Bg1 = (_Float16*)(ws + carve((size_t)KG1_PAD * 128 * 2));
  _Float16* Bg2 = (_Float16*)(ws + carve((size_t)128 * 128 * 2));
  _Float16* Bg3 = (_Float16*)(ws + carve((size_t)128 * 128 * 2));
  _Float16* Bm1 = (_Float16*)(ws + carve((size_t)KM1_REAL * 128 * 2));
  _Float16* Bm2 = (_Float16*)(ws + carve((size_t)128 * 128 * 2));
  _Float16* Bm3 = (_Float16*)(ws + carve((size_t)128 * 128 * 2));
  _Float16* Bm4 = (_Float16*)(ws + carve((size_t)128 * 144 * 2));

  const int TPB = 256;
  auto blocks = [](long long n) { return (unsigned)((n + 255) / 256); };

  // ---- pack weights into fragment-ordered f16 B matrices ----
  pack_weights_k<<<blocks((long long)KG1_PAD * 128), TPB, 0, stream>>>(
      gw1, Bg1, 128, KG1_REAL, KG1_PAD, 128);
  pack_weights_k<<<blocks(128 * 128), TPB, 0, stream>>>(gw2, Bg2, 128, 128, 128, 128);
  pack_weights_k<<<blocks(128 * 128), TPB, 0, stream>>>(gw3, Bg3, 128, 128, 128, 128);
  pack_weights_k<<<blocks((long long)KM1_REAL * 128), TPB, 0, stream>>>(
      mw1, Bm1, 128, KM1_REAL, KM1_REAL, 128);
  pack_weights_k<<<blocks(128 * 128), TPB, 0, stream>>>(mw2, Bm2, 128, 128, 128, 128);
  pack_weights_k<<<blocks(128 * 128), TPB, 0, stream>>>(mw3, Bm3, 128, 128, 128, 128);
  pack_weights_k<<<blocks(128 * 144), TPB, 0, stream>>>(mw4, Bm4, 144, 128, 128, 144);

  // ---- static part of gnet input (x_d3 as f16, channels 5..260) ----
  pack_xd3_k<<<blocks((long long)Bn * FDn * HWn), TPB, 0, stream>>>(x_d3, gin);

  const dim3 gemm_grid128(BHWn / 64, 1);
  const dim3 gemm_grid144(BHWn / 64, (144 + 127) / 128);

  // ---- 3 refinement iterations ----
  for (int it = 0; it < 3; ++it) {
    const float* prev =
        (it == 0) ? ref_gmms : (preds + (size_t)(it - 1) * Bn * 2 * HWn);
    float* predo = preds + (size_t)it * Bn * 2 * HWn;

    cost_volume_k<<<blocks((long long)Bn * Kn * HWn), TPB, 0, stream>>>(
        ref_feat, nghbr_feat, nghbr_gmms, prev, Rs, ts, is_valid, intrins, gin);

    // conv1: 3x3, 261 -> 128, ReLU
    wmma_conv_gemm<3><<<gemm_grid128, TPB, 0, stream>>>(
        gin, GINC * HWn, 0, Bg1, gb1, hA, nullptr,
        KG1_REAL, KG1_PAD, 128, 128, 1);
    // conv2: 1x1, 128 -> 128, ReLU
    wmma_conv_gemm<1><<<gemm_grid128, TPB, 0, stream>>>(
        hA, 128 * HWn, 0, Bg2, gb2, hB, nullptr, 128, 128, 128, 128, 1);
    // conv3: 1x1, 128 -> 128, ReLU
    wmma_conv_gemm<1><<<gemm_grid128, TPB, 0, stream>>>(
        hB, 128 * HWn, 0, Bg3, gb3, hA, nullptr, 128, 128, 128, 128, 1);
    // head: 1x1 128->2 + GMM update
    gnet_head_k<<<blocks(BHWn), TPB, 0, stream>>>(hA, gw4, gb4, prev, predo);
  }

  // ---- mask head (x_d3 view inside gin: channel offset 5, 256 channels) ----
  wmma_conv_gemm<3><<<gemm_grid128, TPB, 0, stream>>>(
      gin, GINC * HWn, Kn, Bm1, mb1, hA, nullptr,
      KM1_REAL, KM1_REAL, 128, 128, 1);
  wmma_conv_gemm<1><<<gemm_grid128, TPB, 0, stream>>>(
      hA, 128 * HWn, 0, Bm2, mb2, hB, nullptr, 128, 128, 128, 128, 1);
  wmma_conv_gemm<1><<<gemm_grid128, TPB, 0, stream>>>(
      hB, 128 * HWn, 0, Bm3, mb3, hA, nullptr, 128, 128, 128, 128, 1);
  // conv4: 1x1, 128 -> 144 logits (no ReLU), f32 output
  wmma_conv_gemm<1><<<gemm_grid144, TPB, 0, stream>>>(
      hA, 128 * HWn, 0, Bm4, mb4, nullptr, maskb, 128, 128, 144, 144, 0);

  // ---- softmax upsample of all 3 predictions into d_out ----
  upsample_k<<<blocks((long long)Bn * HWn * 16), TPB, 0, stream>>>(
      maskb, preds, out);
}
